// GRUModel_6854767804744
// MI455X (gfx1250) — matrix operand order repaired and checked by reading, not compiled, not run on hardware
//
#include <hip/hip_runtime.h>

typedef __attribute__((ext_vector_type(16))) __bf16 v16bf;
typedef __attribute__((ext_vector_type(8)))  float  v8f;
typedef __attribute__((ext_vector_type(4)))  unsigned int u32x4;
typedef __attribute__((ext_vector_type(8)))  int    i32x8;
typedef __attribute__((ext_vector_type(4)))  int    i32x4;

#define TT 512
#define BB 256
#define II 128
#define HH 128
#define GG 384              // 3*H
#define MTOT (TT * BB)      // 131072 rows
#define MTILE 128
#define WBYTES (2 * GG * II * 2)   // both bf16 weight matrices: 196608 B

#if __has_builtin(__builtin_amdgcn_tensor_load_to_lds) && __has_builtin(__builtin_amdgcn_s_wait_tensorcnt)
#define USE_TDM 1
#endif

union Frag16 {
    uint4 u[2];
    v16bf v;
};

__device__ __forceinline__ unsigned short f2bf(float f) {
    unsigned int u = __float_as_uint(f);
    u += 0x7FFFu + ((u >> 16) & 1u);   // round-to-nearest-even
    return (unsigned short)(u >> 16);
}

__device__ __forceinline__ float sigmoidf_(float x) {
    return 1.0f / (1.0f + __expf(-x));
}

__global__ void cvt_weights_kernel(const float* __restrict__ wih,
                                   const float* __restrict__ whh,
                                   unsigned short* __restrict__ out) {
    const int n = GG * II;  // 49152 per matrix
    int i = blockIdx.x * blockDim.x + threadIdx.x;
    if (i < n) {
        out[i]     = f2bf(wih[i]);
        out[n + i] = f2bf(whh[i]);
    }
}

// A fragment: 16x32 bf16, lane L (<16): M=L, K = ks*32+0..15;
// lane L+16: M=L, K = ks*32+16..31.
__device__ __forceinline__ v16bf loadA(const unsigned short* base, int row, int kstep, int lane) {
    const int off = kstep * 32 + (lane >> 4) * 16;
    const uint4* p = (const uint4*)(base + row * II + off);
    Frag16 f;
    f.u[0] = p[0];   // K = off .. off+7
    f.u[1] = p[1];   // K = off+8 .. off+15
    return f.v;
}

// B fragment: 32x16 bf16 of W^T tile. Lane holds N = col + lane%16,
// K = kstep*32 + (lane<16 ? 0..15 : 16..31) -> 32 contiguous bytes of W row.
__device__ __forceinline__ v16bf loadB(const unsigned short* wbase, int col, int kstep, int lane) {
    const int n  = col + (lane & 15);
    const int k0 = kstep * 32 + (lane >> 4) * 16;
    const uint4* p = (const uint4*)(wbase + n * II + k0);
    Frag16 f;
    f.u[0] = p[0];
    f.u[1] = p[1];
    return f.v;
}

__device__ __forceinline__ v8f wmma_bf16(v16bf a, v16bf b, v8f c) {
    return __builtin_amdgcn_wmma_f32_16x16x32_bf16(
        /*neg_a=*/false, a, /*neg_b=*/false, b,
        /*c_mod=*/(short)0, c, /*reuse_a=*/false, /*reuse_b=*/false);
}

__global__ void __launch_bounds__(256, 1)
gru_fused_kernel(const float* __restrict__ x,
                 const float* __restrict__ hid,
                 const unsigned short* __restrict__ wbf,   // [Wih bf16 | Whh bf16]
                 const float* __restrict__ b_ih,
                 const float* __restrict__ b_hh,
                 float* __restrict__ y,
                 float* __restrict__ hout) {
    extern __shared__ __align__(16) char smem[];
    unsigned short* sW = (unsigned short*)smem;                 // 192 KB bf16 weights
    unsigned short* sX = (unsigned short*)(smem + WBYTES);      // 128x128 bf16 = 32 KB
    unsigned short* sH = sX + MTILE * II;                       // 32 KB

    const int mBase = blockIdx.x * MTILE;
    const int tid   = threadIdx.x;

    // CDNA5 prefetch path (global_prefetch_b8) for this block's tile rows.
    __builtin_prefetch(x   + (size_t)(mBase + (tid >> 1)) * II, 0, 0);
    __builtin_prefetch(hid + (size_t)(mBase + (tid >> 1)) * II, 0, 0);

#ifdef USE_TDM
    // Tensor Data Mover: DMA both bf16 weight matrices into LDS as one 2D tile
    // of 4-byte elements: dim0 = 4096 dwords/row, 12 rows = 196608 B.
    if (tid < 32) {
        const unsigned long long ga = (unsigned long long)wbf;
        unsigned ldsa = 0;
#if __has_builtin(__builtin_amdgcn_groupstaticsize)
        ldsa = (unsigned)__builtin_amdgcn_groupstaticsize();  // dynamic-LDS base offset
#endif
        u32x4 g0;
        g0[0] = 1u;                                            // count=1, no gather
        g0[1] = ldsa;                                          // lds_addr
        g0[2] = (unsigned)(ga & 0xFFFFFFFFull);                // global_addr[31:0]
        g0[3] = (unsigned)((ga >> 32) & 0x1FFFFFFull) | (2u << 30);  // addr[56:32] | type=2
        i32x8 g1;
        g1[0] = (int)(2u << 16);      // workgroup_mask=0 (not in cluster), data_size=2 (4B)
        g1[1] = (int)(4096u << 16);   // tensor_dim0[15:0]=4096 (in [31:16])
        g1[2] = (int)(12u << 16);     // tensor_dim0[31:16]=0, tensor_dim1[15:0]=12
        g1[3] = (int)(4096u << 16);   // tensor_dim1[31:16]=0, tile_dim0=4096
        g1[4] = 12;                   // tile_dim1=12, tile_dim2=0
        g1[5] = 4096;                 // tensor_dim0_stride[31:0]=4096
        g1[6] = 0;                    // stride0[47:32]=0, dim1_stride[15:0]=0
        g1[7] = 0;
        i32x4 gz = {0, 0, 0, 0};      // groups 2/3 unused for 2D tensors
#if __clang_major__ >= 23
        i32x8 gz8 = {0, 0, 0, 0, 0, 0, 0, 0};
        __builtin_amdgcn_tensor_load_to_lds(g0, g1, gz, gz, gz8, 0);
#else
        __builtin_amdgcn_tensor_load_to_lds(g0, g1, gz, gz, 0);
#endif
        __builtin_amdgcn_s_wait_tensorcnt(0);
    }
#else
    // Fallback: stage bf16 weights into LDS manually (196608 B = 12288 uint4).
    {
        const uint4* gw = (const uint4*)wbf;
        uint4* lw = (uint4*)sW;
        #pragma unroll 4
        for (int i = tid; i < WBYTES / 16; i += 256) lw[i] = gw[i];
    }
#endif

    // Stage x / hid tiles, converting f32 -> bf16 (float4 per step).
    for (int i = tid; i < MTILE * (II / 4); i += 256) {
        const int row = i >> 5;
        const int c4  = (i & 31) << 2;
        const float4 vx = *(const float4*)(x   + (size_t)(mBase + row) * II + c4);
        const float4 vh = *(const float4*)(hid + (size_t)(mBase + row) * II + c4);
        uint2 px, ph;
        px.x = (unsigned)f2bf(vx.x) | ((unsigned)f2bf(vx.y) << 16);
        px.y = (unsigned)f2bf(vx.z) | ((unsigned)f2bf(vx.w) << 16);
        ph.x = (unsigned)f2bf(vh.x) | ((unsigned)f2bf(vh.y) << 16);
        ph.y = (unsigned)f2bf(vh.z) | ((unsigned)f2bf(vh.w) << 16);
        *(uint2*)(sX + row * II + c4) = px;
        *(uint2*)(sH + row * II + c4) = ph;
    }
    __syncthreads();

    const int lane = tid & 31;
    const int wave = tid >> 5;
    const int nh   = wave;                  // this wave's 16-col slab of H
    const unsigned short* sWih = sW;
    const unsigned short* sWhh = sW + GG * II;

    const int cr = nh * 16;                 // r-gate columns
    const int cz = HH + nh * 16;            // z-gate columns
    const int cn = 2 * HH + nh * 16;        // n-gate columns

    const int c    = lane & 15;
    const int hcol = nh * 16 + c;
    const float br   = b_ih[hcol] + b_hh[hcol];
    const float bz   = b_ih[HH + hcol] + b_hh[HH + hcol];
    const float bi_n = b_ih[2 * HH + hcol];
    const float bh_n = b_hh[2 * HH + hcol];

    for (int msub = 0; msub < 8; ++msub) {
        const int m0l  = msub * 16;
        const int rowA = m0l + (lane & 15);

        v16bf ax[4], ah[4];
        #pragma unroll
        for (int ks = 0; ks < 4; ++ks) {
            ax[ks] = loadA(sX, rowA, ks, lane);
            ah[ks] = loadA(sH, rowA, ks, lane);
        }

        v8f accR = {}, accZ = {}, accIN = {}, accHN = {};
        #pragma unroll
        for (int ks = 0; ks < 4; ++ks) {
            // r and z gates: gi + gh fused into one accumulator each.
            accR  = wmma_bf16(ax[ks], loadB(sWih, cr, ks, lane), accR);
            accR  = wmma_bf16(ah[ks], loadB(sWhh, cr, ks, lane), accR);
            accZ  = wmma_bf16(ax[ks], loadB(sWih, cz, ks, lane), accZ);
            accZ  = wmma_bf16(ah[ks], loadB(sWhh, cz, ks, lane), accZ);
            // n gate: i_n and h_n kept separate (n = tanh(i_n + r*h_n)).
            accIN = wmma_bf16(ax[ks], loadB(sWih, cn, ks, lane), accIN);
            accHN = wmma_bf16(ah[ks], loadB(sWhh, cn, ks, lane), accHN);
        }

        // C/D layout: VGPR v, lanes 0-15 -> M = v; lanes 16-31 -> M = v+8; N = lane%16.
        const int rowHalf = (lane >> 4) << 3;
        #pragma unroll
        for (int v = 0; v < 8; ++v) {
            const int gRow = mBase + m0l + rowHalf + v;
            const size_t oidx = (size_t)gRow * HH + hcol;
            const float r  = sigmoidf_(accR[v] + br);
            const float z  = sigmoidf_(accZ[v] + bz);
            const float nn = tanhf(accIN[v] + bi_n + r * (accHN[v] + bh_n));
            const float hv = hid[oidx];
            y[oidx]    = (1.0f - z) * nn + z * hv;
            hout[oidx] = hv;      // second tuple output: verbatim hid copy, fused
        }
    }
}

extern "C" void kernel_launch(void* const* d_in, const int* in_sizes, int n_in,
                              void* d_out, int out_size, void* d_ws, size_t ws_size,
                              hipStream_t stream) {
    const float* x    = (const float*)d_in[0];
    const float* hid  = (const float*)d_in[1];
    const float* W_ih = (const float*)d_in[2];
    const float* W_hh = (const float*)d_in[3];
    const float* b_ih = (const float*)d_in[4];
    const float* b_hh = (const float*)d_in[5];

    float* y    = (float*)d_out;
    float* hout = y + (size_t)MTOT * HH;
    unsigned short* wbf = (unsigned short*)d_ws;   // 192 KB of scratch used

    // Kernel 0: convert weights to bf16 once per launch (tiny).
    cvt_weights_kernel<<<(GG * II + 255) / 256, 256, 0, stream>>>(W_ih, W_hh, wbf);

    // Kernel 1: fused dual-GEMM + GRU gates. 256 KB dynamic LDS (<= 320 KB/WGP).
    const size_t lds_bytes = WBYTES + 2 * MTILE * II * 2;  // 262144
    (void)hipFuncSetAttribute(reinterpret_cast<const void*>(gru_fused_kernel),
                              hipFuncAttributeMaxDynamicSharedMemorySize,
                              (int)lds_bytes);
    gru_fused_kernel<<<MTOT / MTILE, 256, lds_bytes, stream>>>(
        x, hid, wbf, b_ih, b_hh, y, hout);
    (void)in_sizes; (void)n_in; (void)out_size; (void)ws_size;
}
